// TransitionDown_72129680769552
// MI455X (gfx1250) — compile-verified, matrix operand order checked by reading, not run on hardware
//
#include <hip/hip_runtime.h>
#include <hip/hip_bf16.h>

// TransitionDown (PointNet++ SA layer) for MI455X / gfx1250, wave32 + WMMA.
// B=16, N=4096, K=64, P(NPTS)=1024, FEAT=64, Cin=67->128->128.
//
// FPS (serial, LDS dist array) -> bitonic sort of centroid ids ->
// per-centroid 64-NN via bitonic sort of packed (dist,idx) keys ->
// one-shot f16 weight pre-pack in WMMA B-fragment layout ->
// three fused WMMA GEMM passes that recompute instead of materializing
// 512MB intermediates (BatchNorm needs global batch stats).

#define B_   16
#define N_   4096
#define K_   64
#define P_   1024
#define FEAT_ 64
#define CIN_  67
#define CPAD_ 96     // 3 k-chunks of 32
#define C1_   128
#define C2_   128
#define BN_EPS_ 1e-5f
#define CNT_ 1048576.0f   // B*P*K elements per channel for batch stats

#define WP1_ELEMS_ (8 * 3 * 32 * 16)   // [ntile][ks][lane][16]
#define WP2_ELEMS_ (8 * 4 * 32 * 16)

typedef __attribute__((ext_vector_type(16))) _Float16 v16h;
typedef __attribute__((ext_vector_type(8)))  float    v8f;

// ---------------------------------------------------------------- FPS ------
__global__ __launch_bounds__(256)
void fps_kernel(const float* __restrict__ pos, int* __restrict__ cent) {
    __shared__ float sDist[N_];
    __shared__ float sRedV[256];
    __shared__ int   sRedI[256];
    __shared__ int   sLast;
    const int b = blockIdx.x;
    const int t = threadIdx.x;
    for (int i = t; i < N_; i += 256) sDist[i] = 1e10f;
    if (t == 0) { sLast = 0; cent[b * P_ + 0] = 0; }
    __syncthreads();
    for (int s = 1; s < P_; ++s) {
        const int li = sLast;
        const float px = pos[((size_t)b * N_ + li) * 3 + 0];
        const float py = pos[((size_t)b * N_ + li) * 3 + 1];
        const float pz = pos[((size_t)b * N_ + li) * 3 + 2];
        float bestD = -1.0f; int bestI = 0;
        for (int i = t; i < N_; i += 256) {
            const float dx = pos[((size_t)b * N_ + i) * 3 + 0] - px;
            const float dy = pos[((size_t)b * N_ + i) * 3 + 1] - py;
            const float dz = pos[((size_t)b * N_ + i) * 3 + 2] - pz;
            float d = dx * dx + dy * dy + dz * dz;
            d = fminf(sDist[i], d);
            sDist[i] = d;
            if (d > bestD) { bestD = d; bestI = i; }  // strict > => first max
        }
        sRedV[t] = bestD; sRedI[t] = bestI;
        __syncthreads();
        for (int off = 128; off > 0; off >>= 1) {
            if (t < off) {
                const float v2 = sRedV[t + off]; const int i2 = sRedI[t + off];
                if (v2 > sRedV[t] || (v2 == sRedV[t] && i2 < sRedI[t])) {
                    sRedV[t] = v2; sRedI[t] = i2;
                }
            }
            __syncthreads();
        }
        if (t == 0) { sLast = sRedI[0]; cent[b * P_ + s] = sRedI[0]; }
        __syncthreads();
    }
}

// ------------------------------------------------- sort centroid ids -------
__global__ __launch_bounds__(256)
void sortcent_kernel(int* __restrict__ cent) {
    __shared__ int s[P_];
    const int b = blockIdx.x, t = threadIdx.x;
    for (int i = t; i < P_; i += 256) s[i] = cent[b * P_ + i];
    __syncthreads();
    for (int k = 2; k <= P_; k <<= 1)
        for (int j = k >> 1; j > 0; j >>= 1) {
            for (int i = t; i < P_; i += 256) {
                const int ixj = i ^ j;
                if (ixj > i) {
                    const bool up = (i & k) == 0;
                    const int a = s[i], c = s[ixj];
                    if ((a > c) == up) { s[i] = c; s[ixj] = a; }
                }
            }
            __syncthreads();
        }
    for (int i = t; i < P_; i += 256) cent[b * P_ + i] = s[i];
}

// ------------------------------------------------------------- 64-NN -------
__global__ __launch_bounds__(256)
void knn_kernel(const float* __restrict__ pos, const int* __restrict__ cent,
                int* __restrict__ nidx) {
    __shared__ unsigned long long keys[N_];   // (dist_bits << 32) | idx
    const int bp = blockIdx.x;
    const int b = bp >> 10, p = bp & (P_ - 1);
    const int t = threadIdx.x;
    const int ci = cent[b * P_ + p];
    const float cx = pos[((size_t)b * N_ + ci) * 3 + 0];
    const float cy = pos[((size_t)b * N_ + ci) * 3 + 1];
    const float cz = pos[((size_t)b * N_ + ci) * 3 + 2];
    for (int i = t; i < N_; i += 256) {
        const float dx = pos[((size_t)b * N_ + i) * 3 + 0] - cx;
        const float dy = pos[((size_t)b * N_ + i) * 3 + 1] - cy;
        const float dz = pos[((size_t)b * N_ + i) * 3 + 2] - cz;
        const float d = dx * dx + dy * dy + dz * dz;  // >= 0 -> bits monotonic
        keys[i] = ((unsigned long long)__float_as_uint(d) << 32) | (unsigned)i;
    }
    __syncthreads();
    for (int k = 2; k <= N_; k <<= 1)
        for (int j = k >> 1; j > 0; j >>= 1) {
            for (int i = t; i < N_; i += 256) {
                const int ixj = i ^ j;
                if (ixj > i) {
                    const bool up = (i & k) == 0;
                    const unsigned long long a = keys[i], c = keys[ixj];
                    if ((a > c) == up) { keys[i] = c; keys[ixj] = a; }
                }
            }
            __syncthreads();
        }
    if (t < K_)
        nidx[(size_t)bp * K_ + t] = (int)(keys[t] & 0xffffffffu);
}

// --------------------------------------- f16 weight pre-pack (B layout) ----
// Fragment layout per ISA 16-bit B 32x16: lane ln<16 holds col n, K =
// k0+{0..7,16..23}; lane ln+16 holds K = k0+{8..15,24..31}.  Packed order:
// wp[((ntile*NKS + ks)*32 + lane)*16 + j]  -> one aligned 32B load per lane.
__global__ __launch_bounds__(256)
void pack_weights_kernel(const float* __restrict__ W1,
                         const float* __restrict__ W2,
                         _Float16* __restrict__ wp1,
                         _Float16* __restrict__ wp2) {
    const int gid = blockIdx.x * 256 + threadIdx.x;
    if (gid < WP1_ELEMS_) {
        const int j = gid & 15, lane = (gid >> 4) & 31;
        const int rest = gid >> 9, ks = rest % 3, ntile = rest / 3;
        const int n = ntile * 16 + (lane & 15);
        const int kk = ks * 32 + (lane >> 4) * 8 + (j & 7) + ((j >> 3) << 4);
        wp1[gid] = (kk < CIN_) ? (_Float16)W1[n * CIN_ + kk] : (_Float16)0.0f;
    }
    if (gid < WP2_ELEMS_) {
        const int j = gid & 15, lane = (gid >> 4) & 31;
        const int ks = (gid >> 9) & 3, ntile = gid >> 11;
        const int n = ntile * 16 + (lane & 15);
        const int kk = ks * 32 + (lane >> 4) * 8 + (j & 7) + ((j >> 3) << 4);
        wp2[gid] = (_Float16)W2[n * C1_ + kk];
    }
}

// ------------------------------------------------------- small helpers -----
__global__ void zero_stats_kernel(float* s1, float* s2) {
    const int t = threadIdx.x;
    if (t < 256) { s1[t] = 0.0f; s2[t] = 0.0f; }
}

__global__ void bn_finalize_kernel(const float* __restrict__ stats,
                                   const float* __restrict__ g,
                                   const float* __restrict__ bt,
                                   float* __restrict__ scsh) {
    const int c = threadIdx.x;
    if (c < 128) {
        const float mu = stats[c] / CNT_;
        const float var = stats[128 + c] / CNT_ - mu * mu;
        const float sc = g[c] * rsqrtf(var + BN_EPS_);
        scsh[c] = sc;
        scsh[128 + c] = bt[c] - mu * sc;
    }
}

// ------------------------------------------------- WMMA fragment loaders ---
__device__ inline v16h load_a_lds(const _Float16* base, int row, int ldk,
                                  int k0, int half) {
    v16h a;
#pragma unroll
    for (int j = 0; j < 8; ++j) {
        a[j]     = base[row * ldk + k0 + half * 8 + j];        // K = k0+{0..15}
        a[j + 8] = base[row * ldk + k0 + 16 + half * 8 + j];   // K = k0+{16..31}
    }
    return a;
}

__device__ inline v16h load_b_pack(const _Float16* __restrict__ wp,
                                   int ntile, int nks, int ks, int lane) {
    return *(const v16h*)(wp + (((size_t)(ntile * nks + ks) * 32 + lane) << 4));
}

// ------------------------------------------------------ fused GEMM passes --
// PHASE 0: gather + GEMM1                      -> atomic stats of y1
// PHASE 1: gather + GEMM1 + BN1/ReLU + GEMM2   -> atomic stats of y2
// PHASE 2: full chain + BN2/ReLU + maxpool(K)  -> cpos + new_feat
template <int PHASE>
__global__ __launch_bounds__(256)
void fused_kernel(const float* __restrict__ pos, const float* __restrict__ feat,
                  const _Float16* __restrict__ wp1, const _Float16* __restrict__ wp2,
                  const int* __restrict__ cent, const int* __restrict__ nidx,
                  const float* __restrict__ scsh1, const float* __restrict__ scsh2,
                  float* __restrict__ stats_out, float* __restrict__ out) {
    __shared__ _Float16 hA[K_ * CPAD_];   // gathered [dpos|feat] rows, f16
    __shared__ _Float16 h1[K_ * C1_];     // layer-1 activations, f16
    __shared__ int      nbr[K_];
    __shared__ float    cp[3];
    __shared__ float    sSum[128], sSq[128];
    __shared__ unsigned sMax[128];

    const int bp = blockIdx.x;
    const int b = bp >> 10, p = bp & (P_ - 1);
    const int t = threadIdx.x;

    if (t < K_) nbr[t] = nidx[(size_t)bp * K_ + t];
    if (t < 3)  cp[t]  = pos[((size_t)b * N_ + cent[b * P_ + p]) * 3 + t];
    if (t < 128) { sSum[t] = 0.0f; sSq[t] = 0.0f; sMax[t] = 0u; }
    __syncthreads();

    // gather: feat rows via coalesced float4 loads (cols 3..66 of hA)
    for (int idx = t; idx < K_ * (FEAT_ / 4); idx += 256) {
        const int r = idx >> 4, c4 = (idx & 15) << 2;
        const float4 f = *(const float4*)(feat + ((size_t)b * N_ + nbr[r]) * FEAT_ + c4);
        _Float16* dst = &hA[r * CPAD_ + 3 + c4];
        dst[0] = (_Float16)f.x; dst[1] = (_Float16)f.y;
        dst[2] = (_Float16)f.z; dst[3] = (_Float16)f.w;
    }
    // gather: pos deltas (cols 0..2) + zero pad (cols 67..95)
    for (int idx = t; idx < K_ * 32; idx += 256) {
        const int r = idx >> 5, c = idx & 31;
        if (c < 3)
            hA[r * CPAD_ + c] = (_Float16)(pos[((size_t)b * N_ + nbr[r]) * 3 + c] - cp[c]);
        else
            hA[r * CPAD_ + 64 + c] = (_Float16)0.0f;   // cols 67..95
    }
    __syncthreads();

    const int wave = t >> 5, lane = t & 31;
    const int half = lane >> 4, ln = lane & 15;
    const int mt = wave & 3;              // 16-row strip (4 strips * 16 = 64)
    const int ng = (wave >> 2) * 4;       // 4 col tiles per wave (2 groups)
    const int row = mt * 16 + ln;

    // ---------------- layer 1: y1[64x128] = hA[64x96] * W1^T ----------------
    v8f acc[4] = {{0,0,0,0,0,0,0,0}, {0,0,0,0,0,0,0,0},
                  {0,0,0,0,0,0,0,0}, {0,0,0,0,0,0,0,0}};
#pragma unroll
    for (int ks = 0; ks < 3; ++ks) {
        const int k0 = ks * 32;
        const v16h a = load_a_lds(hA, row, CPAD_, k0, half);
#pragma unroll
        for (int nt = 0; nt < 4; ++nt) {
            const v16h bf = load_b_pack(wp1, ng + nt, 3, ks, lane);
            acc[nt] = __builtin_amdgcn_wmma_f32_16x16x32_f16(
                false, a, false, bf, (short)0, acc[nt], false, false);
        }
    }

    if constexpr (PHASE == 0) {
        // batch-stat partial sums of pre-BN y1
#pragma unroll
        for (int nt = 0; nt < 4; ++nt) {
            const int n = (ng + nt) * 16 + ln;
            float s = 0.0f, q = 0.0f;
#pragma unroll
            for (int j = 0; j < 8; ++j) { s += acc[nt][j]; q += acc[nt][j] * acc[nt][j]; }
            atomicAdd(&sSum[n], s);
            atomicAdd(&sSq[n], q);
        }
        __syncthreads();
        if (t < 128) {
            atomicAdd(&stats_out[t], sSum[t]);
            atomicAdd(&stats_out[128 + t], sSq[t]);
        }
        return;
    } else {
        // BN1 + ReLU -> LDS h1 (f16)
#pragma unroll
        for (int nt = 0; nt < 4; ++nt) {
            const int n = (ng + nt) * 16 + ln;
            const float sc = scsh1[n], sh = scsh1[128 + n];
#pragma unroll
            for (int j = 0; j < 8; ++j) {
                const int m = mt * 16 + j + half * 8;   // C-layout: M=j / j+8
                const float v = fmaxf(acc[nt][j] * sc + sh, 0.0f);
                h1[m * C1_ + n] = (_Float16)v;
            }
        }
        __syncthreads();

        // -------------- layer 2: y2[64x128] = h1[64x128] * W2^T -------------
        v8f acc2[4] = {{0,0,0,0,0,0,0,0}, {0,0,0,0,0,0,0,0},
                       {0,0,0,0,0,0,0,0}, {0,0,0,0,0,0,0,0}};
#pragma unroll
        for (int ks = 0; ks < 4; ++ks) {
            const int k0 = ks * 32;
            const v16h a = load_a_lds(h1, row, C1_, k0, half);
#pragma unroll
            for (int nt = 0; nt < 4; ++nt) {
                const v16h bf = load_b_pack(wp2, ng + nt, 4, ks, lane);
                acc2[nt] = __builtin_amdgcn_wmma_f32_16x16x32_f16(
                    false, a, false, bf, (short)0, acc2[nt], false, false);
            }
        }

        if constexpr (PHASE == 1) {
#pragma unroll
            for (int nt = 0; nt < 4; ++nt) {
                const int n = (ng + nt) * 16 + ln;
                float s = 0.0f, q = 0.0f;
#pragma unroll
                for (int j = 0; j < 8; ++j) { s += acc2[nt][j]; q += acc2[nt][j] * acc2[nt][j]; }
                atomicAdd(&sSum[n], s);
                atomicAdd(&sSq[n], q);
            }
            __syncthreads();
            if (t < 128) {
                atomicAdd(&stats_out[t], sSum[t]);
                atomicAdd(&stats_out[128 + t], sSq[t]);
            }
            return;
        } else {
            // BN2 + ReLU + maxpool over the 64 neighbor rows
#pragma unroll
            for (int nt = 0; nt < 4; ++nt) {
                const int n = (ng + nt) * 16 + ln;
                const float sc = scsh2[n], sh = scsh2[128 + n];
                float lm = 0.0f;   // ReLU output >= 0
#pragma unroll
                for (int j = 0; j < 8; ++j)
                    lm = fmaxf(lm, fmaxf(acc2[nt][j] * sc + sh, 0.0f));
                // non-negative floats: uint bit-pattern order == float order
                atomicMax(&sMax[n], __float_as_uint(lm));
            }
            __syncthreads();
            if (t < 128)
                out[(size_t)B_ * P_ * 3 + (size_t)bp * C2_ + t] = __uint_as_float(sMax[t]);
            if (t < 3)
                out[(size_t)bp * 3 + t] = cp[t];
        }
    }
}

// ------------------------------------------------------------- launcher ----
extern "C" void kernel_launch(void* const* d_in, const int* in_sizes, int n_in,
                              void* d_out, int out_size, void* d_ws, size_t ws_size,
                              hipStream_t stream) {
    const float* pos  = (const float*)d_in[0];
    const float* feat = (const float*)d_in[1];
    const float* W1   = (const float*)d_in[2];
    // d_in[3] (b1) and d_in[7] (b2): conv bias cancels exactly under
    // batch-stat BatchNorm (mean absorbs it) -> skipped.
    const float* g1   = (const float*)d_in[4];
    const float* bt1  = (const float*)d_in[5];
    const float* W2   = (const float*)d_in[6];
    const float* g2   = (const float*)d_in[8];
    const float* bt2  = (const float*)d_in[9];
    float* out = (float*)d_out;

    char* ws = (char*)d_ws;
    size_t off = 0;
    int* cent = (int*)(ws + off);              off += (size_t)B_ * P_ * 4;
    int* nidx = (int*)(ws + off);              off += (size_t)B_ * P_ * K_ * 4;
    float* stats1 = (float*)(ws + off);        off += 256 * 4;
    float* scsh1  = (float*)(ws + off);        off += 256 * 4;
    float* stats2 = (float*)(ws + off);        off += 256 * 4;
    float* scsh2  = (float*)(ws + off);        off += 256 * 4;
    off = (off + 63) & ~(size_t)63;            // 64B align for v16h loads
    _Float16* wp1 = (_Float16*)(ws + off);     off += (size_t)WP1_ELEMS_ * 2;
    off = (off + 63) & ~(size_t)63;
    _Float16* wp2 = (_Float16*)(ws + off);     off += (size_t)WP2_ELEMS_ * 2;

    fps_kernel<<<B_, 256, 0, stream>>>(pos, cent);
    sortcent_kernel<<<B_, 256, 0, stream>>>(cent);
    knn_kernel<<<B_ * P_, 256, 0, stream>>>(pos, cent, nidx);
    pack_weights_kernel<<<(WP2_ELEMS_ + 255) / 256, 256, 0, stream>>>(W1, W2, wp1, wp2);
    zero_stats_kernel<<<1, 256, 0, stream>>>(stats1, stats2);
    fused_kernel<0><<<B_ * P_, 256, 0, stream>>>(pos, feat, wp1, wp2, cent, nidx,
                                                 nullptr, nullptr, stats1, nullptr);
    bn_finalize_kernel<<<1, 128, 0, stream>>>(stats1, g1, bt1, scsh1);
    fused_kernel<1><<<B_ * P_, 256, 0, stream>>>(pos, feat, wp1, wp2, cent, nidx,
                                                 scsh1, nullptr, stats2, nullptr);
    bn_finalize_kernel<<<1, 128, 0, stream>>>(stats2, g2, bt2, scsh2);
    fused_kernel<2><<<B_ * P_, 256, 0, stream>>>(pos, feat, wp1, wp2, cent, nidx,
                                                 scsh1, scsh2, nullptr, out);
}
